// TensorField_64914135711932
// MI455X (gfx1250) — matrix operand order repaired and checked by reading, not compiled, not run on hardware
//
#include <hip/hip_runtime.h>
#include <hip/hip_bf16.h>

// ---------------------------------------------------------------------------
// TensorField Equiformer block for MI455X (gfx1250, wave32, WMMA 16x16x32 f16)
// - all GEMMs (incl. attention logits) on v_wmma_f32_16x16x32_f16
// - weights pre-transposed to f16 so B fragments are 2x ds_load_b128 per tile
// - W tiles staged into LDS via global_load_async_to_lds_b128 (ASYNCcnt) and
//   the Tensor Data Mover (tensor_load_to_lds + s_wait_tensorcnt)
// - xv gather rows warmed with global_prefetch
// ---------------------------------------------------------------------------

typedef __attribute__((ext_vector_type(16))) _Float16 v16h;
typedef __attribute__((ext_vector_type(8)))  float    v8f;
typedef __attribute__((ext_vector_type(4)))  unsigned int u32x4;
typedef __attribute__((ext_vector_type(8)))  int      i32x8;
typedef __attribute__((ext_vector_type(4)))  int      i32x4;

#define DEV __device__ __forceinline__

constexpr int NQ = 4096, NS = 16384, NBATCH = 4;
constexpr int DIM = 128, NHEAD = 8, KNN = 32;
constexpr int TDIM = 32, ESDIM = 96;         // edge-scalar width = 64 + 32
constexpr int CAND = NS / NBATCH;            // same-batch candidates per query
constexpr float R_MAX = 1.0f, R_MIN_NS = 0.1f;
constexpr float PI_F   = 3.14159265358979f;
constexpr float LOG1E4 = 9.210340371976184f; // log(10000)
constexpr float C1 = 1.7320508075688772f;    // sqrt(3)
constexpr float C2 = 3.872983346207417f;     // sqrt(15)
constexpr float C3 = 1.118033988749895f;     // sqrt(5)/2

DEV float silu_f(float x) { return x / (1.0f + __expf(-x)); }

// --- CDNA5 async global->LDS copy (GLOBAL_LOAD_ASYNC_TO_LDS_B128, ASYNCcnt)
DEV void async_g2l_b128(void* lds_ptr, const void* gptr) {
  // flat shared pointer: addr[31:0] is the LDS byte address (ISA 10.2 aperture)
  unsigned int loff = (unsigned int)(unsigned long long)(uintptr_t)lds_ptr;
  unsigned long long ga = (unsigned long long)(uintptr_t)gptr;
  asm volatile("global_load_async_to_lds_b128 %0, %1, off"
               :: "v"(loff), "v"(ga) : "memory");
}
DEV void wait_async0() {
  asm volatile("s_wait_asynccnt 0x0" ::: "memory");
}

// --- CDNA5 Tensor Data Mover: 1-D contiguous f16 tile, global -> LDS -------
#if __has_builtin(__builtin_amdgcn_tensor_load_to_lds)
DEV void tdm_load_1d_f16(const void* gptr, void* lds_ptr, unsigned nelem) {
  unsigned long long ga = (unsigned long long)(uintptr_t)gptr;
  unsigned lo = (unsigned)(uintptr_t)lds_ptr;
  u32x4 g0;
  g0.x = 1u;                                             // count=1, user D#
  g0.y = lo;                                             // lds_addr (bytes)
  g0.z = (unsigned)ga;                                   // global_addr[31:0]
  g0.w = (unsigned)((ga >> 32) & 0x01FFFFFFull) | (2u << 30); // addr[56:32]|type=2
  i32x8 g1;
  g1[0] = (int)(1u << 16);                               // data_size=2 bytes
  g1[1] = (int)((nelem & 0xFFFFu) << 16);                // tensor_dim0[15:0]
  g1[2] = (int)(((nelem >> 16) & 0xFFFFu) | (1u << 16)); // dim0 hi | tensor_dim1=1
  g1[3] = (int)((nelem & 0xFFFFu) << 16);                // tile_dim0 = nelem
  g1[4] = 1;                                             // tile_dim1 = 1
  g1[5] = (int)nelem;                                    // tensor_dim0_stride
  g1[6] = 0; g1[7] = 0;
  i32x4 z4 = {0, 0, 0, 0};
#if defined(__clang_major__) && (__clang_major__ >= 23)
  i32x8 z8 = {};
  __builtin_amdgcn_tensor_load_to_lds(g0, g1, z4, z4, z8, 0);
#else
  __builtin_amdgcn_tensor_load_to_lds(g0, g1, z4, z4, 0);
#endif
}
#endif

// --- WMMA fragment loaders (ISA 7.12.2, wave32) ----------------------------
// A: 16x32 f16, row-major LDS. Lane's 16 halves are one contiguous run at K = kb + 16*(lane>>4).
DEV v16h load_A_frag(const _Float16* lds, int r0, int kb, int ld, int lane) {
  int row  = r0 + (lane & 15);
  int half = lane >> 4;
  const _Float16* p = lds + row * ld + kb;
  v16h a;
#pragma unroll
  for (int e = 0; e < 16; ++e) {
    int k = e + 16 * half;
    a[e] = p[k];
  }
  return a;
}
// B: 32x16 f16 from TRANSPOSED weights WT[n][k] (ld = K). Lane owns column
// n0+(lane&15), K = kb + 16*(lane>>4) + e  -> 16 contiguous halves (32 bytes).
DEV v16h load_B_fragT(const _Float16* ldsT, int kb, int n0, int ldk, int lane) {
  const _Float16* p = ldsT + (n0 + (lane & 15)) * ldk + kb + (lane >> 4) * 16;
  v16h b;
#pragma unroll
  for (int e = 0; e < 16; ++e) b[e] = p[e];
  return b;
}

DEV v8f wmma_f16(v16h a, v16h b, v8f c) {
  return __builtin_amdgcn_wmma_f32_16x16x32_f16(false, a, false, b,
                                                (short)0, c, false, false);
}

// ---------------------------------------------------------------------------
// Kernel: fp32 [K,N] -> fp16 transposed [N,K] weight preconversion (tiny)
// ---------------------------------------------------------------------------
__global__ void cvt_f16_T_kernel(const float* __restrict__ src,
                                 _Float16* __restrict__ dst,
                                 int rows, int cols) {
  int i = blockIdx.x * blockDim.x + threadIdx.x;
  if (i < rows * cols) {
    int r = i / cols, c = i % cols;
    dst[c * rows + r] = (_Float16)src[i];
  }
}

// ---------------------------------------------------------------------------
// Kernel: K-nearest same-batch neighbors. One workgroup per query.
// ---------------------------------------------------------------------------
__global__ __launch_bounds__(256) void knn_kernel(
    const float* __restrict__ qc, const float* __restrict__ ic,
    const int* __restrict__ qb_arr, const int* __restrict__ ib_arr,
    int* __restrict__ idx_out) {
  __shared__ float sD[CAND];
  __shared__ float rv[256];
  __shared__ int   ri[256];
  int q = blockIdx.x, tid = threadIdx.x;
  int qb = qb_arr[q];
  float qx = qc[q * 3 + 0], qy = qc[q * 3 + 1], qz = qc[q * 3 + 2];
  int base = qb * CAND;   // batches are contiguous equal blocks
  for (int i = tid; i < CAND; i += 256) {
    int j = base + i;
    float d2 = 1e30f;
    if (ib_arr[j] == qb) {
      float dx = ic[j * 3 + 0] - qx;
      float dy = ic[j * 3 + 1] - qy;
      float dz = ic[j * 3 + 2] - qz;
      d2 = dx * dx + dy * dy + dz * dz;
    }
    sD[i] = d2;
  }
  __syncthreads();
  for (int kk = 0; kk < KNN; ++kk) {
    float best = 1e30f; int bi = -1;
    for (int i = tid; i < CAND; i += 256)
      if (sD[i] < best) { best = sD[i]; bi = i; }
    rv[tid] = best; ri[tid] = bi;
    __syncthreads();
    for (int s = 128; s > 0; s >>= 1) {
      if (tid < s) {
        float ov = rv[tid + s]; int oi = ri[tid + s];
        if (ov < rv[tid] ||
            (ov == rv[tid] && oi >= 0 && (ri[tid] < 0 || oi < ri[tid]))) {
          rv[tid] = ov; ri[tid] = oi;
        }
      }
      __syncthreads();
    }
    if (tid == 0) {
      int w = ri[0];
      idx_out[q * KNN + kk] = base + (w < 0 ? 0 : w);
      if (w >= 0) sD[w] = 1e30f;   // remove winner
    }
    __syncthreads();
  }
}

// ---------------------------------------------------------------------------
// Kernel: [M,128]@[128,128] f16 WMMA GEMM (W transposed), f32 out.
// SKIP is a compile-time flag -> branch-free epilogue with batched loads.
// Block = 128 threads (4 waves); covers 64 rows; W staged by async LDS copy.
// ---------------------------------------------------------------------------
template <int SKIP>
__global__ __launch_bounds__(128) void gemm128_f16_kernel(
    const float* __restrict__ A, const _Float16* __restrict__ WT,
    float* __restrict__ out) {
  alignas(16) __shared__ _Float16 sA[64 * DIM];
  alignas(16) __shared__ _Float16 sW[DIM * DIM];   // transposed [N][K]
  int tid = threadIdx.x;
  int row0 = blockIdx.x * 64;
  // async DMA of the f16 weight panel into LDS (16 B per lane per issue)
  for (int c = tid; c < DIM * DIM / 8; c += 128)
    async_g2l_b128(&sW[c * 8], WT + c * 8);
  for (int i = tid; i < 64 * DIM; i += 128)
    sA[i] = (_Float16)A[(size_t)row0 * DIM + i];
  wait_async0();
  __syncthreads();
  int wave = tid >> 5, lane = tid & 31;
  int mrow = wave * 16;
  // A fragments are N-invariant: load all four K-chunks once.
  v16h a[4];
#pragma unroll
  for (int kc = 0; kc < 4; ++kc)
    a[kc] = load_A_frag(sA, mrow, kc * 32, DIM, lane);
  for (int nt = 0; nt < 8; ++nt) {
    // issue all B fragment loads before the WMMA burst
    v16h b[4];
#pragma unroll
    for (int kc = 0; kc < 4; ++kc)
      b[kc] = load_B_fragT(sW, kc * 32, nt * 16, DIM, lane);
    v8f acc = {};
#pragma unroll
    for (int kc = 0; kc < 4; ++kc) acc = wmma_f16(a[kc], b[kc], acc);
    int col = nt * 16 + (lane & 15);
    int rbase = mrow + 8 * (lane >> 4);
    float skipv[8];
    if (SKIP) {
#pragma unroll
      for (int r = 0; r < 8; ++r)
        skipv[r] = A[(size_t)(row0 + rbase + r) * DIM + col];
    }
#pragma unroll
    for (int r = 0; r < 8; ++r) {
      float v = acc[r];
      if (SKIP) v += skipv[r];
      out[(size_t)(row0 + rbase + r) * DIM + col] = v;
    }
  }
}

// ---------------------------------------------------------------------------
// Kernel: fused per-query edge pipeline. One workgroup (4 waves) per query.
// 32 edges = 2 WMMA M-tiles; entire MLP chain stays in LDS.
// ---------------------------------------------------------------------------
__global__ __launch_bounds__(128) void tf_fused_kernel(
    const float* __restrict__ query_coord, const float* __restrict__ input_coord,
    const float* __restrict__ time_emb,
    const int* __restrict__ query_b, const int* __restrict__ input_b,
    const int* __restrict__ idxbuf, const float* __restrict__ xv,
    const _Float16* __restrict__ w1t, const float* __restrict__ b1,
    const _Float16* __restrict__ w2t, const float* __restrict__ b2,
    const _Float16* __restrict__ wr2vt, const _Float16* __restrict__ wat,
    const float* __restrict__ W_sh, float* __restrict__ agg) {
  alignas(16) __shared__ _Float16 sES[KNN][ESDIM];  // 6 KB edge scalars
  alignas(16) __shared__ _Float16 sW[DIM * DIM];    // 32 KB: W1T/W2T/Wr2vT
  alignas(16) __shared__ _Float16 sR[KNN][DIM];     // 8 KB rad
  alignas(16) __shared__ _Float16 sV[KNN][DIM];     // 8 KB modulated values
  alignas(16) __shared__ _Float16 sWAT[16][DIM];    // 4 KB W_alpha^T, N pad 16
  __shared__ _Float16 sWsh[9 * DIM];                // 2.25 KB W_sh
  __shared__ float sSH[KNN][9];
  __shared__ float sAlpha[KNN][NHEAD];              // logits, then softmax'd
  __shared__ float sLen[KNN];
  __shared__ int   sIdx[KNN];
  __shared__ int   sValid[KNN];

  int q = blockIdx.x, tid = threadIdx.x;
  int wave = tid >> 5, lane = tid & 31;
  int qb = query_b[q];

  // kick off async DMA of W1T (f16, [128][96]) into LDS right away
  for (int c = tid; c < ESDIM * DIM / 8; c += 128)
    async_g2l_b128(&sW[c * 8], w1t + c * 8);

  // ---- phase 0: edge geometry (one thread per edge) + small weight copies
  if (tid < KNN) {
    int j = idxbuf[q * KNN + tid];
    __builtin_prefetch(xv + (size_t)j * DIM, 0, 0);   // warm gather rows
    float vx = input_coord[j * 3 + 0] - query_coord[q * 3 + 0];
    float vy = input_coord[j * 3 + 1] - query_coord[q * 3 + 1];
    float vz = input_coord[j * 3 + 2] - query_coord[q * 3 + 2];
    float len = __builtin_sqrtf(vx * vx + vy * vy + vz * vz + 1e-12f);
    int valid = (len < R_MAX) && (input_b[j] == qb);
    float t = fminf(len / R_MAX, 1.0f);
    float w_max = 0.5f * (__cosf(PI_F * t) + 1.0f);
    float w_s = valid ? w_max : 0.0f;
    float w_min = (len < R_MIN_NS)
                      ? (0.5f - 0.5f * __cosf(PI_F * len / R_MIN_NS)) : 1.0f;
    float w_ns = w_s * w_min;
    float inv = 1.0f / len;
    float x = vx * inv, y = vy * inv, z = vz * inv;
    sSH[tid][0] = w_s;                       // l=0 * scalar cutoff
    sSH[tid][1] = C1 * x * w_ns;
    sSH[tid][2] = C1 * y * w_ns;
    sSH[tid][3] = C1 * z * w_ns;
    sSH[tid][4] = C2 * x * y * w_ns;
    sSH[tid][5] = C2 * y * z * w_ns;
    sSH[tid][6] = C3 * (3.0f * z * z - 1.0f) * w_ns;
    sSH[tid][7] = C2 * x * z * w_ns;
    sSH[tid][8] = (C2 * 0.5f) * (x * x - y * y) * w_ns;
    sIdx[tid] = j; sLen[tid] = len; sValid[tid] = valid;
  }
  for (int i = tid; i < 16 * DIM; i += 128) {          // W_alpha^T, zero-padded
    int n = i >> 7;
    sWAT[0][i] = (n < NHEAD) ? wat[(size_t)n * DIM + (i & 127)] : (_Float16)0.0f;
  }
  for (int i = tid; i < 9 * DIM; i += 128) sWsh[i] = (_Float16)W_sh[i];
  __syncthreads();

  // ---- phase 1: edge scalars (sinusoidal ++ time emb)
  {
    int e = tid >> 2, c0 = (tid & 3) * 24;
    float len = sLen[e];
#pragma unroll
    for (int c = 0; c < 24; ++c) {
      int col = c0 + c;
      float v;
      if (col < 32) {
        v = __sinf(len * __expf(-LOG1E4 * (float)col * (1.0f / 31.0f)));
      } else if (col < 64) {
        v = __cosf(len * __expf(-LOG1E4 * (float)(col - 32) * (1.0f / 31.0f)));
      } else {
        v = time_emb[qb * TDIM + (col - 64)];
      }
      sES[e][col] = (_Float16)v;
    }
  }
  wait_async0();          // W1T resident in LDS
  __syncthreads();

  // ---- GEMM1: rad1 = silu(ES @ W1 + b1), M=32 K=96 N=128
  {
    v8f acc[2][2] = {};
    for (int kc = 0; kc < 3; ++kc) {
      v16h a0 = load_A_frag(&sES[0][0], 0, kc * 32, ESDIM, lane);
      v16h a1 = load_A_frag(&sES[0][0], 16, kc * 32, ESDIM, lane);
      v16h bb0 = load_B_fragT(sW, kc * 32, (wave * 2 + 0) * 16, ESDIM, lane);
      v16h bb1 = load_B_fragT(sW, kc * 32, (wave * 2 + 1) * 16, ESDIM, lane);
      acc[0][0] = wmma_f16(a0, bb0, acc[0][0]);
      acc[1][0] = wmma_f16(a1, bb0, acc[1][0]);
      acc[0][1] = wmma_f16(a0, bb1, acc[0][1]);
      acc[1][1] = wmma_f16(a1, bb1, acc[1][1]);
    }
#pragma unroll
    for (int mt = 0; mt < 2; ++mt)
#pragma unroll
      for (int j = 0; j < 2; ++j) {
        int col = (wave * 2 + j) * 16 + (lane & 15);
        int e0 = mt * 16 + 8 * (lane >> 4);
        float bb = b1[col];
#pragma unroll
        for (int r = 0; r < 8; ++r)
          sR[e0 + r][col] = (_Float16)silu_f(acc[mt][j][r] + bb);
      }
  }
  __syncthreads();         // all waves done with W1T reads and sR writes
  // ---- stage W2T via the Tensor Data Mover (TENSORcnt) ----
#if __has_builtin(__builtin_amdgcn_tensor_load_to_lds)
  if (wave == 0) {
    tdm_load_1d_f16(w2t, sW, DIM * DIM);
    __builtin_amdgcn_s_wait_tensorcnt(0);
  }
#else
  for (int i = tid; i < DIM * DIM; i += 128) sW[i] = w2t[i];
#endif
  __syncthreads();

  // ---- GEMM2: rad = silu(rad1 @ W2 + b2), M=32 K=128 N=128
  {
    v8f acc[2][2] = {};
    for (int kc = 0; kc < 4; ++kc) {
      v16h a0 = load_A_frag(&sR[0][0], 0, kc * 32, DIM, lane);
      v16h a1 = load_A_frag(&sR[0][0], 16, kc * 32, DIM, lane);
      v16h bb0 = load_B_fragT(sW, kc * 32, (wave * 2 + 0) * 16, DIM, lane);
      v16h bb1 = load_B_fragT(sW, kc * 32, (wave * 2 + 1) * 16, DIM, lane);
      acc[0][0] = wmma_f16(a0, bb0, acc[0][0]);
      acc[1][0] = wmma_f16(a1, bb0, acc[1][0]);
      acc[0][1] = wmma_f16(a0, bb1, acc[0][1]);
      acc[1][1] = wmma_f16(a1, bb1, acc[1][1]);
    }
    __syncthreads();   // everyone done reading sR/sW before overwrite
#pragma unroll
    for (int mt = 0; mt < 2; ++mt)
#pragma unroll
      for (int j = 0; j < 2; ++j) {
        int col = (wave * 2 + j) * 16 + (lane & 15);
        int e0 = mt * 16 + 8 * (lane >> 4);
        float bb = b2[col];
#pragma unroll
        for (int r = 0; r < 8; ++r)
          sR[e0 + r][col] = (_Float16)silu_f(acc[mt][j][r] + bb);
      }
  }
  // ---- stage W_rad2val^T via TDM while the epilogue finishes ----
#if __has_builtin(__builtin_amdgcn_tensor_load_to_lds)
  if (wave == 0) {
    tdm_load_1d_f16(wr2vt, sW, DIM * DIM);
    __builtin_amdgcn_s_wait_tensorcnt(0);
  }
#else
  for (int i = tid; i < DIM * DIM; i += 128) sW[i] = wr2vt[i];
#endif
  __syncthreads();

  // ---- GEMM3: rv = rad @ W_rad2val;  v = xv[idx] * rv * (sh @ W_sh)
  {
    v8f acc[2][2] = {};
    for (int kc = 0; kc < 4; ++kc) {
      v16h a0 = load_A_frag(&sR[0][0], 0, kc * 32, DIM, lane);
      v16h a1 = load_A_frag(&sR[0][0], 16, kc * 32, DIM, lane);
      v16h bb0 = load_B_fragT(sW, kc * 32, (wave * 2 + 0) * 16, DIM, lane);
      v16h bb1 = load_B_fragT(sW, kc * 32, (wave * 2 + 1) * 16, DIM, lane);
      acc[0][0] = wmma_f16(a0, bb0, acc[0][0]);
      acc[1][0] = wmma_f16(a1, bb0, acc[1][0]);
      acc[0][1] = wmma_f16(a0, bb1, acc[0][1]);
      acc[1][1] = wmma_f16(a1, bb1, acc[1][1]);
    }
#pragma unroll
    for (int mt = 0; mt < 2; ++mt)
#pragma unroll
      for (int j = 0; j < 2; ++j) {
        int col = (wave * 2 + j) * 16 + (lane & 15);
        int e0 = mt * 16 + 8 * (lane >> 4);
#pragma unroll
        for (int r = 0; r < 8; ++r) {
          int e = e0 + r;
          float shw = 0.0f;
#pragma unroll
          for (int l = 0; l < 9; ++l)
            shw += sSH[e][l] * (float)sWsh[l * DIM + col];
          float xvv = xv[(size_t)sIdx[e] * DIM + col];
          sV[e][col] = (_Float16)(xvv * acc[mt][j][r] * shw);
        }
      }
  }
  __syncthreads();

  // ---- logits = rad @ W_alpha via WMMA (N padded to 16), wave 0 only
  if (wave == 0) {
    v8f acc0 = {}, acc1 = {};
    for (int kc = 0; kc < 4; ++kc) {
      v16h a0 = load_A_frag(&sR[0][0], 0, kc * 32, DIM, lane);
      v16h a1 = load_A_frag(&sR[0][0], 16, kc * 32, DIM, lane);
      v16h b  = load_B_fragT(&sWAT[0][0], kc * 32, 0, DIM, lane);
      acc0 = wmma_f16(a0, b, acc0);
      acc1 = wmma_f16(a1, b, acc1);
    }
    int col = lane & 15;
    if (col < NHEAD) {
      int ebase = 8 * (lane >> 4);
#pragma unroll
      for (int r = 0; r < 8; ++r) {
        int e0 = ebase + r;
        sAlpha[e0][col] = sValid[e0] ? acc0[r] : -1e9f;
        int e1 = 16 + ebase + r;
        sAlpha[e1][col] = sValid[e1] ? acc1[r] : -1e9f;
      }
    }
  }
  __syncthreads();

  // ---- softmax over the 32 neighbors, per head (in place in sAlpha)
  if (tid < NHEAD) {
    int h = tid;
    float mx = -1e30f;
    for (int e = 0; e < KNN; ++e) mx = fmaxf(mx, sAlpha[e][h]);
    float sum = 0.0f;
    for (int e = 0; e < KNN; ++e) {
      float ex = __expf(sAlpha[e][h] - mx);
      sAlpha[e][h] = ex; sum += ex;
    }
    float inv = 1.0f / sum;
    for (int e = 0; e < KNN; ++e) sAlpha[e][h] *= inv;
  }
  __syncthreads();

  // ---- aggregate: agg[d] = sum_e alpha[e, d/16] * v[e, d]
  {
    int d = tid, h = d >> 4;   // HD = 16
    float s = 0.0f;
#pragma unroll
    for (int e = 0; e < KNN; ++e) s += sAlpha[e][h] * (float)sV[e][d];
    agg[(size_t)q * DIM + d] = s;
  }
}

// ---------------------------------------------------------------------------
extern "C" void kernel_launch(void* const* d_in, const int* in_sizes, int n_in,
                              void* d_out, int out_size, void* d_ws,
                              size_t ws_size, hipStream_t stream) {
  (void)in_sizes; (void)n_in; (void)out_size; (void)ws_size;
  const float* query_x     = (const float*)d_in[0]; (void)query_x;
  const float* query_coord = (const float*)d_in[1];
  const float* input_x     = (const float*)d_in[2];
  const float* input_coord = (const float*)d_in[3];
  const float* time_emb    = (const float*)d_in[4];
  const float* W1      = (const float*)d_in[5];
  const float* b1      = (const float*)d_in[6];
  const float* W2      = (const float*)d_in[7];
  const float* b2      = (const float*)d_in[8];
  const float* W_val   = (const float*)d_in[9];
  const float* W_r2v   = (const float*)d_in[10];
  const float* W_sh    = (const float*)d_in[11];
  const float* W_alpha = (const float*)d_in[12];
  const float* W_o     = (const float*)d_in[13];
  const int* query_b   = (const int*)d_in[14];
  const int* input_b   = (const int*)d_in[15];

  // workspace layout (~11.2 MB)
  char* ws = (char*)d_ws;
  float* xv  = (float*)ws;       ws += (size_t)NS * DIM * sizeof(float);
  float* agg = (float*)ws;       ws += (size_t)NQ * DIM * sizeof(float);
  int*   idx = (int*)ws;         ws += (size_t)NQ * KNN * sizeof(int);
  _Float16* w1t   = (_Float16*)ws; ws += (size_t)ESDIM * DIM * 2; // [128][96]
  _Float16* w2t   = (_Float16*)ws; ws += (size_t)DIM * DIM * 2;   // [N][K]
  _Float16* wr2vt = (_Float16*)ws; ws += (size_t)DIM * DIM * 2;
  _Float16* wvalt = (_Float16*)ws; ws += (size_t)DIM * DIM * 2;
  _Float16* wot   = (_Float16*)ws; ws += (size_t)DIM * DIM * 2;
  _Float16* wat   = (_Float16*)ws; ws += (size_t)NHEAD * DIM * 2; // [8][128]

  // 1) weights -> f16, transposed to [N][K] (once per launch; tiny)
  cvt_f16_T_kernel<<<(ESDIM * DIM + 255) / 256, 256, 0, stream>>>(W1, w1t, ESDIM, DIM);
  cvt_f16_T_kernel<<<(DIM * DIM + 255) / 256, 256, 0, stream>>>(W2, w2t, DIM, DIM);
  cvt_f16_T_kernel<<<(DIM * DIM + 255) / 256, 256, 0, stream>>>(W_r2v, wr2vt, DIM, DIM);
  cvt_f16_T_kernel<<<(DIM * DIM + 255) / 256, 256, 0, stream>>>(W_val, wvalt, DIM, DIM);
  cvt_f16_T_kernel<<<(DIM * DIM + 255) / 256, 256, 0, stream>>>(W_o, wot, DIM, DIM);
  cvt_f16_T_kernel<<<(DIM * NHEAD + 255) / 256, 256, 0, stream>>>(W_alpha, wat, DIM, NHEAD);

  // 2) K-nearest neighbors per query
  knn_kernel<<<NQ, 256, 0, stream>>>(query_coord, input_coord, query_b,
                                     input_b, idx);

  // 3) xv = input_x @ W_val  (node-level, 8x less work than per-edge)
  gemm128_f16_kernel<0><<<NS / 64, 128, 0, stream>>>(input_x, wvalt, xv);

  // 4) fused per-query edge pipeline (4 WMMA GEMMs + softmax + aggregation)
  tf_fused_kernel<<<NQ, 128, 0, stream>>>(query_coord, input_coord, time_emb,
                                          query_b, input_b, idx, xv,
                                          w1t, b1, w2t, b2, wr2vt, wat,
                                          W_sh, agg);

  // 5) out = agg @ W_o + agg
  gemm128_f16_kernel<1><<<NQ / 64, 128, 0, stream>>>(agg, wot, (float*)d_out);
}
